// SelfAttention_35373350650227
// MI455X (gfx1250) — matrix operand order repaired and checked by reading, not compiled, not run on hardware
//
#include <hip/hip_runtime.h>

// Problem constants (match reference): B=4, N=4096, D=512
#define BB 4
#define NN 4096
#define DD 512
#define KSTEP 32          // keys processed per tile step
#define DH 256            // D columns per attention pass (gridDim.y = 2)

typedef __attribute__((ext_vector_type(16))) __bf16 v16bf;
typedef __attribute__((ext_vector_type(8)))  __bf16 v8bf;
typedef __attribute__((ext_vector_type(4)))  __bf16 v4bf;
typedef __attribute__((ext_vector_type(8)))  float  v8f;
typedef __attribute__((ext_vector_type(4)))  unsigned int v4u;
typedef __attribute__((ext_vector_type(8)))  int v8i;
typedef __attribute__((ext_vector_type(4)))  int v4i;

// Use the Tensor Data Mover when the device pass has the builtins.
#if defined(__HIP_DEVICE_COMPILE__) && __has_builtin(__builtin_amdgcn_tensor_load_to_lds) && __has_builtin(__builtin_amdgcn_s_wait_tensorcnt)
#define USE_TDM 1
#else
#define USE_TDM 0
#endif

// ---------------------------------------------------------------------------
// Kernel 1: f32 -> bf16 conversion (grid covers n/4 float4 chunks)
// ---------------------------------------------------------------------------
__global__ void cvt_f32_bf16_kernel(const float4* __restrict__ src,
                                    __bf16* __restrict__ dst, int n4) {
  int i = blockIdx.x * blockDim.x + threadIdx.x;
  if (i < n4) {
    float4 v = src[i];
    v4bf o;
    o[0] = (__bf16)v.x; o[1] = (__bf16)v.y;
    o[2] = (__bf16)v.z; o[3] = (__bf16)v.w;
    *(v4bf*)(dst + (size_t)i * 4) = o;   // single b64 store
  }
}

// ---------------------------------------------------------------------------
// Kernel 2: QKV projection.  out = X @ W^T + bias  (torch Linear semantics)
// One wave computes one 16x16 output tile via 16 chained bf16 WMMAs (K=512).
// ---------------------------------------------------------------------------
__global__ void qkv_proj_kernel(const __bf16* __restrict__ Abf,
                                const __bf16* __restrict__ Bbf,
                                const __bf16* __restrict__ Wqb,
                                const __bf16* __restrict__ Wkb,
                                const __bf16* __restrict__ Wvb,
                                const float* __restrict__ bq,
                                const float* __restrict__ bk,
                                const float* __restrict__ bv,
                                __bf16* __restrict__ Qb,
                                __bf16* __restrict__ Kb,
                                __bf16* __restrict__ Vb) {
  const int lane = threadIdx.x & 31;
  const int w    = blockIdx.x * 8 + (threadIdx.x >> 5);
  const int COLT = DD / 16;                         // 32
  const int TPM  = (BB * NN / 16) * COLT;           // 32768 tiles / matrix
  const int mat  = w / TPM;
  const int rem  = w % TPM;
  const int rowT = rem / COLT, colT = rem % COLT;
  const int half = lane >> 4, hl = lane & 15;

  const __bf16* X    = (mat == 2) ? Bbf : Abf;
  const __bf16* W    = (mat == 0) ? Wqb : (mat == 1) ? Wkb : Wvb;
  const float*  bias = (mat == 0) ? bq  : (mat == 1) ? bk  : bv;
  __bf16*       Out  = (mat == 0) ? Qb  : (mat == 1) ? Kb  : Vb;

  const int row0 = rowT * 16, col0 = colT * 16;
  const __bf16* xrow = X + (size_t)(row0 + hl) * DD;
  const __bf16* wrow = W + (size_t)(col0 + hl) * DD;

  v8f acc = {0.f, 0.f, 0.f, 0.f, 0.f, 0.f, 0.f, 0.f};
#pragma unroll
  for (int kk = 0; kk < DD; kk += 32) {
    v8bf a0 = *(const v8bf*)(xrow + kk + half * 16);
    v8bf a1 = *(const v8bf*)(xrow + kk + half * 16 + 8);
    v16bf a;
#pragma unroll
    for (int i = 0; i < 8; ++i) { a[i] = a0[i]; a[i + 8] = a1[i]; }
    v16bf bm = *(const v16bf*)(wrow + kk + half * 16);
    acc = __builtin_amdgcn_wmma_f32_16x16x32_bf16(false, a, false, bm,
                                                  (short)0, acc, false, false);
  }
  const float bval = bias[col0 + hl];
#pragma unroll
  for (int r = 0; r < 8; ++r) {
    const int row = row0 + r + 8 * half;
    Out[(size_t)row * DD + col0 + hl] = (__bf16)(acc[r] + bval);
  }
}

// ---------------------------------------------------------------------------
// TDM descriptor: load one 32x512 bf16 K tile (contiguous rows) into LDS.
//   D# group0: count=1 | lds_addr | global_addr(57b) | type=2
//   D# group1: data_size=2B, tensor 512x32, tile 512x32, dim0 stride 512
// Issued by one wave; completion tracked with TENSORcnt.
// ---------------------------------------------------------------------------
#if USE_TDM
__device__ inline void tdm_load_k_tile(const __bf16* gptr, void* lds_ptr) {
  const unsigned long long ga = (unsigned long long)(uintptr_t)gptr;
  const unsigned lds_off = (unsigned)(uintptr_t)lds_ptr;  // LDS aperture: addr[31:0]
  v4u g0;
  g0[0] = 1u;                                        // count=1, user mode
  g0[1] = lds_off;                                   // lds_addr
  g0[2] = (unsigned)(ga & 0xffffffffu);              // global_addr[31:0]
  g0[3] = (unsigned)((ga >> 32) & 0x1ffffffu)        // global_addr[56:32]
        | (2u << 30);                                // type=2 ("image")
  v8i g1;
  g1[0] = (int)(1u << 16);       // workgroup_mask=0, data_size=1 (2 bytes)
  g1[1] = (int)(512u << 16);     // tensor_dim0[15:0] = 512
  g1[2] = (int)(32u << 16);      // tensor_dim0 hi=0, tensor_dim1[15:0] = 32
  g1[3] = (int)(512u << 16);     // tensor_dim1 hi=0, tile_dim0 = 512
  g1[4] = (int)32;               // tile_dim1 = 32, tile_dim2 = 0
  g1[5] = (int)512;              // tensor_dim0_stride[31:0] = 512
  g1[6] = 0;                     // stride hi, tensor_dim1_stride lo
  g1[7] = 0;
  const v4i z4  = {0, 0, 0, 0};                 // groups 2/3 unused (2D tensor)
  const v8i z8  = {0, 0, 0, 0, 0, 0, 0, 0};     // trailing group (clang-23 form)
  __builtin_amdgcn_tensor_load_to_lds(g0, g1, z4, z4, z8, 0);
}
#endif

// ---------------------------------------------------------------------------
// Kernel 3: flash attention, double-buffered K (TDM) + V (manual transpose).
// Block = 256 threads = 8 waves; each wave owns 16 Q rows; gridDim.y = D half.
// Per 32-key step: 32 WMMAs for S = Q K^T, streaming softmax, P fixup in
// per-wave LDS (wave-ordered, no WG barrier), 16 WMMAs for O += P V.
// ---------------------------------------------------------------------------
__global__ void attn_kernel(const __bf16* __restrict__ Qb,
                            const __bf16* __restrict__ Kb,
                            const __bf16* __restrict__ Vb,
                            float* __restrict__ out) {
  extern __shared__ __align__(16) char smem[];
  __bf16* Kt = (__bf16*)smem;                                       // [2][32][512]
  __bf16* Vt = (__bf16*)(smem + 2 * KSTEP * DD * 2);                // [2][256][32]
  __bf16* Pb = (__bf16*)(smem + 2 * KSTEP * DD * 2 + 2 * DH * KSTEP * 2); // [8][16][32]

  const int tid  = threadIdx.x;
  const int lane = tid & 31, wv = tid >> 5;
  const int half = lane >> 4, hl = lane & 15;

  const int rblksPerB = NN / 128;                 // 32
  const int b     = blockIdx.x / rblksPerB;
  const int rblk  = blockIdx.x % rblksPerB;
  const int d0    = blockIdx.y * DH;
  const int qrow0 = rblk * 128 + wv * 16;
  const size_t base = (size_t)b * NN;

  const float scale = 0.044194173824159216f;      // 1/sqrt(512)

  float mrow[8], lrow[8];
  v8f o[16];
#pragma unroll
  for (int r = 0; r < 8; ++r) { mrow[r] = -3.0e38f; lrow[r] = 0.f; }
#pragma unroll
  for (int t = 0; t < 16; ++t)
#pragma unroll
    for (int r = 0; r < 8; ++r) o[t][r] = 0.f;

  const __bf16* qrow_ptr = Qb + (base + qrow0 + hl) * DD;
  __bf16* Pw = Pb + wv * 16 * KSTEP;

  const int NSTEP = NN / KSTEP;                   // 128

  // ---- prologue: stage step 0 into buffer 0 ----
#if USE_TDM
  if (wv == 0) tdm_load_k_tile(Kb + base * DD, Kt);
#else
  {
    const uint4* src = (const uint4*)(Kb + base * DD);
    uint4* dst = (uint4*)Kt;
    for (int i = tid; i < (KSTEP * DD * 2) / 16; i += 256) dst[i] = src[i];
  }
#endif
  for (int i = tid; i < KSTEP * DH; i += 256) {
    const int k = i >> 8, j = i & (DH - 1);
    Vt[j * KSTEP + k] = Vb[(base + k) * DD + d0 + j];
  }
#if USE_TDM
  __builtin_amdgcn_s_wait_tensorcnt(0);
#endif
  __syncthreads();

  for (int s = 0; s < NSTEP; ++s) {
    const int cur = s & 1, nxt = cur ^ 1;
    const int key0 = s * KSTEP;
    const __bf16* Ktc = Kt + cur * (KSTEP * DD);
    const __bf16* Vtc = Vt + cur * (DH * KSTEP);

    // ---- issue next-step staging; overlaps with this step's compute ----
    if (s + 1 < NSTEP) {
      const int knext = key0 + KSTEP;
#if USE_TDM
      if (wv == 0)
        tdm_load_k_tile(Kb + (base + knext) * DD, Kt + nxt * (KSTEP * DD));
#else
      {
        const uint4* src = (const uint4*)(Kb + (base + knext) * DD);
        uint4* dst = (uint4*)(Kt + nxt * (KSTEP * DD));
        for (int i = tid; i < (KSTEP * DD * 2) / 16; i += 256) dst[i] = src[i];
      }
#endif
      __bf16* Vtn = Vt + nxt * (DH * KSTEP);
      for (int i = tid; i < KSTEP * DH; i += 256) {
        const int k = i >> 8, j = i & (DH - 1);
        Vtn[j * KSTEP + k] = Vb[(base + knext + k) * DD + d0 + j];
      }
    }

    // ---- S = Q K^T for two 16-key subtiles, accumulate over D=512 ----
    v8f s0 = {0.f,0.f,0.f,0.f,0.f,0.f,0.f,0.f};
    v8f s1 = {0.f,0.f,0.f,0.f,0.f,0.f,0.f,0.f};
#pragma unroll
    for (int dt = 0; dt < 16; ++dt) {
      v8bf a0 = *(const v8bf*)(qrow_ptr + dt * 32 + half * 16);
      v8bf a1 = *(const v8bf*)(qrow_ptr + dt * 32 + half * 16 + 8);
      v16bf a;
#pragma unroll
      for (int i = 0; i < 8; ++i) { a[i] = a0[i]; a[i + 8] = a1[i]; }
      v16bf k0 = *(const v16bf*)(Ktc + (size_t)hl        * DD + dt * 32 + half * 16);
      v16bf k1 = *(const v16bf*)(Ktc + (size_t)(16 + hl) * DD + dt * 32 + half * 16);
      s0 = __builtin_amdgcn_wmma_f32_16x16x32_bf16(false, a, false, k0,
                                                   (short)0, s0, false, false);
      s1 = __builtin_amdgcn_wmma_f32_16x16x32_bf16(false, a, false, k1,
                                                   (short)0, s1, false, false);
    }

    // ---- streaming softmax update (row = r + 8*half, cols across 16 lanes)
    float alpha[8];
#pragma unroll
    for (int r = 0; r < 8; ++r) {
      s0[r] *= scale; s1[r] *= scale;
      float v = fmaxf(s0[r], s1[r]);
      v = fmaxf(v, __shfl_xor(v, 1));
      v = fmaxf(v, __shfl_xor(v, 2));
      v = fmaxf(v, __shfl_xor(v, 4));
      v = fmaxf(v, __shfl_xor(v, 8));
      const float mn = fmaxf(mrow[r], v);
      alpha[r] = __expf(mrow[r] - mn);
      mrow[r]  = mn;
      const float p0 = __expf(s0[r] - mn);
      const float p1 = __expf(s1[r] - mn);
      s0[r] = p0; s1[r] = p1;
      float ps = p0 + p1;
      ps += __shfl_xor(ps, 1);
      ps += __shfl_xor(ps, 2);
      ps += __shfl_xor(ps, 4);
      ps += __shfl_xor(ps, 8);
      lrow[r] = alpha[r] * lrow[r] + ps;
    }

    // ---- C-layout -> A-fragment layout fixup for P via per-wave LDS.
    // LDS ops within a wave are in-order (DScnt); wave_barrier stops the
    // compiler from reordering the reads above the writes.
#pragma unroll
    for (int r = 0; r < 8; ++r) {
      Pw[(r + 8 * half) * KSTEP + hl]      = (__bf16)s0[r];
      Pw[(r + 8 * half) * KSTEP + 16 + hl] = (__bf16)s1[r];
    }
    __builtin_amdgcn_wave_barrier();
    v8bf p0 = *(const v8bf*)(Pw + hl * KSTEP + half * 16);
    v8bf p1 = *(const v8bf*)(Pw + hl * KSTEP + half * 16 + 8);
    v16bf pa;
#pragma unroll
    for (int i = 0; i < 8; ++i) { pa[i] = p0[i]; pa[i + 8] = p1[i]; }
    __builtin_amdgcn_wave_barrier();

    // ---- O = O*alpha + P @ V  (16 d-tiles of 16 cols) ----
#pragma unroll
    for (int dt = 0; dt < 16; ++dt) {
      v16bf bv = *(const v16bf*)(Vtc + (dt * 16 + hl) * KSTEP + half * 16);
#pragma unroll
      for (int r = 0; r < 8; ++r) o[dt][r] *= alpha[r];
      o[dt] = __builtin_amdgcn_wmma_f32_16x16x32_bf16(false, pa, false, bv,
                                                      (short)0, o[dt], false, false);
    }

    // ---- close the step: next-step DMA done, everyone done with cur ----
#if USE_TDM
    __builtin_amdgcn_s_wait_tensorcnt(0);
#endif
    __syncthreads();
  }

  // ---- epilogue: normalize and store f32 ----
#pragma unroll
  for (int r = 0; r < 8; ++r) {
    const float inv = 1.f / lrow[r];
    const int row = qrow0 + r + 8 * half;
#pragma unroll
    for (int dt = 0; dt < 16; ++dt)
      out[(base + row) * DD + d0 + dt * 16 + hl] = o[dt][r] * inv;
  }
}

// ---------------------------------------------------------------------------
// Launcher
// ---------------------------------------------------------------------------
extern "C" void kernel_launch(void* const* d_in, const int* in_sizes, int n_in,
                              void* d_out, int out_size, void* d_ws, size_t ws_size,
                              hipStream_t stream) {
  const float* in_a = (const float*)d_in[0];
  const float* in_b = (const float*)d_in[1];
  const float* Wq   = (const float*)d_in[2];
  const float* bq   = (const float*)d_in[3];
  const float* Wk   = (const float*)d_in[4];
  const float* bk   = (const float*)d_in[5];
  const float* Wv   = (const float*)d_in[6];
  const float* bv   = (const float*)d_in[7];
  float* out = (float*)d_out;

  const size_t nIn = (size_t)BB * NN * DD;   // 8,388,608
  const size_t nW  = (size_t)DD * DD;        // 262,144

  char* ws = (char*)d_ws;
  __bf16* Abf = (__bf16*)ws; ws += nIn * 2;
  __bf16* Bbf = (__bf16*)ws; ws += nIn * 2;
  __bf16* Wqb = (__bf16*)ws; ws += nW * 2;
  __bf16* Wkb = (__bf16*)ws; ws += nW * 2;
  __bf16* Wvb = (__bf16*)ws; ws += nW * 2;
  __bf16* Qb  = (__bf16*)ws; ws += nIn * 2;
  __bf16* Kb  = (__bf16*)ws; ws += nIn * 2;
  __bf16* Vb  = (__bf16*)ws; ws += nIn * 2;

  // 1) f32 -> bf16 conversions
  {
    int n4 = (int)(nIn / 4);
    int g  = (n4 + 255) / 256;
    cvt_f32_bf16_kernel<<<g, 256, 0, stream>>>((const float4*)in_a, Abf, n4);
    cvt_f32_bf16_kernel<<<g, 256, 0, stream>>>((const float4*)in_b, Bbf, n4);
    int w4 = (int)(nW / 4);
    int gw = (w4 + 255) / 256;
    cvt_f32_bf16_kernel<<<gw, 256, 0, stream>>>((const float4*)Wq, Wqb, w4);
    cvt_f32_bf16_kernel<<<gw, 256, 0, stream>>>((const float4*)Wk, Wkb, w4);
    cvt_f32_bf16_kernel<<<gw, 256, 0, stream>>>((const float4*)Wv, Wvb, w4);
  }

  // 2) QKV projections: 3 * 1024 * 32 = 98304 tiles, 8 waves per block
  qkv_proj_kernel<<<98304 / 8, 256, 0, stream>>>(Abf, Bbf, Wqb, Wkb, Wvb,
                                                 bq, bk, bv, Qb, Kb, Vb);

  // 3) flash attention: 128 row-blocks x 2 D-halves, 104 KB dynamic LDS
  dim3 grid(BB * NN / 128, 2);
  size_t shmem = (size_t)2 * KSTEP * DD * 2      // K tiles  64 KB (double buffered)
               + (size_t)2 * DH * KSTEP * 2      // V^T tiles 32 KB (double buffered)
               + (size_t)8 * 16 * KSTEP * 2;     // P buffer   8 KB
  attn_kernel<<<grid, 256, shmem, stream>>>(Qb, Kb, Vb, out);
}
